// ContrastivePuzzleLoss_68015102099702
// MI455X (gfx1250) — matrix operand order repaired and checked by reading, not compiled, not run on hardware
//
#include <hip/hip_runtime.h>
#include <math.h>

typedef __attribute__((ext_vector_type(16))) _Float16 v16h;
typedef __attribute__((ext_vector_type(8)))  _Float16 v8h;
typedef __attribute__((ext_vector_type(4)))  _Float16 v4h;
typedef __attribute__((ext_vector_type(8)))  float    v8f;
typedef __attribute__((ext_vector_type(4)))  int      v4i;

#define AS1 __attribute__((address_space(1)))
#define AS3 __attribute__((address_space(3)))

#define B_  64
#define N_  576
#define D_  768
#define IT  64                       // anchor rows per block
#define JC  64                       // J columns per staged chunk
#define KC  128                      // K elements per staged chunk
#define SROW 136                     // LDS row stride in halves (128 + 8 pad)
#define NJC (N_ / JC)                // 9
#define NKC (D_ / KC)                // 6
#define NBLK_GRAM (B_ * (N_ / IT))   // 576
#define INV_TEMP (1.0f / 0.07f)

// --- gfx1250 async global->LDS copy (16B per lane), with safe fallbacks ----
__device__ __forceinline__ void async_copy16(const _Float16* g, _Float16* l) {
#if __has_builtin(__builtin_amdgcn_global_load_async_to_lds_b128)
  __builtin_amdgcn_global_load_async_to_lds_b128(
      (AS1 v4i*)g, (AS3 v4i*)l, 0, 0);
#else
  *(v8h*)l = *(const v8h*)g;
#endif
}
__device__ __forceinline__ void async_wait0() {
#if __has_builtin(__builtin_amdgcn_global_load_async_to_lds_b128)
#if __has_builtin(__builtin_amdgcn_s_wait_asynccnt)
  __builtin_amdgcn_s_wait_asynccnt(0);
#else
  asm volatile("s_wait_asynccnt 0x0" ::: "memory");
#endif
#endif
}

// ---------------------------------------------------------------------------
// P1 kernel 1: normalize + convert, f̂[row] = f16(features[row] / max(||.||,eps))
// One wave per row; also the only pass that touches fp32 features.
// ---------------------------------------------------------------------------
__global__ __launch_bounds__(256) void cpl_normconv_kernel(
    const float* __restrict__ feat, _Float16* __restrict__ fhat) {
  const int wave = threadIdx.x >> 5;
  const int lane = threadIdx.x & 31;
  const int row  = blockIdx.x * 8 + wave;
  const float* p = feat + (size_t)row * D_;
  float4 q[6];
  float  s = 0.0f;
#pragma unroll
  for (int t = 0; t < 6; ++t) {
    q[t] = *(const float4*)(p + lane * 4 + t * 128);
    s += q[t].x * q[t].x + q[t].y * q[t].y + q[t].z * q[t].z + q[t].w * q[t].w;
  }
#pragma unroll
  for (int m = 16; m >= 1; m >>= 1) s += __shfl_xor(s, m, 32);
  const float inv = 1.0f / fmaxf(sqrtf(s), 1e-12f);
  _Float16* dst = fhat + (size_t)row * D_;
#pragma unroll
  for (int t = 0; t < 6; ++t) {
    v4h h;
    h[0] = (_Float16)(q[t].x * inv); h[1] = (_Float16)(q[t].y * inv);
    h[2] = (_Float16)(q[t].z * inv); h[3] = (_Float16)(q[t].w * inv);
    *(v4h*)(dst + lane * 4 + t * 128) = h;
  }
}

// ---------------------------------------------------------------------------
// P1 kernel 2: Gram from pre-converted f16 with async LDS staging.
// Steady state: async b128 stage -> 6 ds_load_b128 -> 2 v_wmma per K-step.
// ---------------------------------------------------------------------------
__global__ __launch_bounds__(256) void cpl_gram_loss_f16_kernel(
    const _Float16* __restrict__ fhat,
    const int*      __restrict__ pos32,
    float* __restrict__ partial) {
  __shared__ _Float16 As[IT * SROW];
  __shared__ _Float16 Bs[JC * SROW];
  __shared__ int      iposs[IT];
  __shared__ float    rowacc[2 * IT];

  const int b    = blockIdx.x / (N_ / IT);
  const int i0   = (blockIdx.x % (N_ / IT)) * IT;
  const int tid  = threadIdx.x;
  const int wv   = tid >> 5;
  const int lane = tid & 31;
  const int isub = wv & 3;
  const int jsel = wv >> 2;
  const int l15  = lane & 15;
  const int h    = lane >> 4;

  if (tid < IT) iposs[tid] = pos32[2 * (b * N_ + i0 + tid)];

  const int sr = tid >> 2;           // staging row 0..63
  const int sc = (tid & 3) * 32;     // staging col slice

  __syncthreads();

  int prow[8];
#pragma unroll
  for (int v = 0; v < 8; ++v) prow[v] = iposs[isub * 16 + 8 * h + v];
  const int igBase = i0 + isub * 16 + 8 * h;

  float racc[8];
#pragma unroll
  for (int v = 0; v < 8; ++v) racc[v] = 0.0f;

  for (int jc = 0; jc < NJC; ++jc) {
    const int j0  = jc * JC;
    const int jg0 = j0 + jsel * 32 + l15;
    const int jg1 = jg0 + 16;
    const int pj0 = pos32[2 * (b * N_ + jg0)];
    const int pj1 = pos32[2 * (b * N_ + jg1)];

    v8f acc0 = {};
    v8f acc1 = {};

    for (int kc = 0; kc < NKC; ++kc) {
      __syncthreads();   // previous consumers done before restaging
      {
        const _Float16* srcA = fhat + (size_t)(b * N_ + i0 + sr) * D_ + kc * KC + sc;
        const _Float16* srcB = fhat + (size_t)(b * N_ + j0 + sr) * D_ + kc * KC + sc;
        _Float16* dstA = &As[sr * SROW + sc];
        _Float16* dstB = &Bs[sr * SROW + sc];
#pragma unroll
        for (int t = 0; t < 4; ++t) {
          async_copy16(srcA + t * 8, dstA + t * 8);
          async_copy16(srcB + t * 8, dstB + t * 8);
        }
      }
      async_wait0();
      __syncthreads();

#pragma unroll
      for (int ks = 0; ks < 4; ++ks) {
        const int kb = ks * 32;
        v16h a;
        {
          const _Float16* ar = &As[(isub * 16 + l15) * SROW + kb];
          const v8h c0 = *(const v8h*)(ar + 8 * h);
          const v8h c1 = *(const v8h*)(ar + 16 + 8 * h);
#pragma unroll
          for (int e = 0; e < 8; ++e) { a[e] = c0[e]; a[8 + e] = c1[e]; }
        }
        v16h b0f, b1f;
        {
          const _Float16* br0 = &Bs[(jsel * 32 + l15) * SROW + kb + 8 * h];
          const v8h d0 = *(const v8h*)(br0);
          const v8h d1 = *(const v8h*)(br0 + 16);
          const _Float16* br1 = &Bs[(jsel * 32 + 16 + l15) * SROW + kb + 8 * h];
          const v8h e0 = *(const v8h*)(br1);
          const v8h e1 = *(const v8h*)(br1 + 16);
#pragma unroll
          for (int e = 0; e < 8; ++e) {
            b0f[e] = d0[e]; b0f[8 + e] = d1[e];
            b1f[e] = e0[e]; b1f[8 + e] = e1[e];
          }
        }
        acc0 = __builtin_amdgcn_wmma_f32_16x16x32_f16(
            false, a, false, b0f, (short)0, acc0, false, false);
        acc1 = __builtin_amdgcn_wmma_f32_16x16x32_f16(
            false, a, false, b1f, (short)0, acc1, false, false);
      }
    }

#pragma unroll
    for (int v = 0; v < 8; ++v) {
      const int ig = igBase + v;
      float w0 = (prow[v] == pj0) ? -1.0f : 1.0f;
      w0 = (ig == jg0) ? 0.0f : w0;
      float w1 = (prow[v] == pj1) ? -1.0f : 1.0f;
      w1 = (ig == jg1) ? 0.0f : w1;
      racc[v] = fmaf(acc0[v], w0, racc[v]);
      racc[v] = fmaf(acc1[v], w1, racc[v]);
    }
  }

#pragma unroll
  for (int v = 0; v < 8; ++v) {
#pragma unroll
    for (int m = 8; m >= 1; m >>= 1) racc[v] += __shfl_xor(racc[v], m, 32);
  }
  if (l15 == 0) {
#pragma unroll
    for (int v = 0; v < 8; ++v)
      rowacc[jsel * IT + isub * 16 + 8 * h + v] = racc[v];
  }
  __syncthreads();

  if (tid < IT) {
    const float acc = rowacc[tid] + rowacc[IT + tid];
    const float x   = acc * INV_TEMP;
    const float lp  = (x > 0.0f) ? (x + log1pf(expf(-x))) : log1pf(expf(x));
    rowacc[tid] = lp;
  }
  __syncthreads();
  if (tid == 0) {
    float s = 0.0f;
    for (int r = 0; r < IT; ++r) s += rowacc[r];
    partial[blockIdx.x] = s;
  }
}

// ---------------------------------------------------------------------------
// P2 kernel 1 (fallback, small ws): inv_norm only.
// ---------------------------------------------------------------------------
__global__ __launch_bounds__(256) void cpl_norms_kernel(
    const float* __restrict__ feat, float* __restrict__ invn) {
  const int wave = threadIdx.x >> 5;
  const int lane = threadIdx.x & 31;
  const int row  = blockIdx.x * 8 + wave;
  const float* p = feat + (size_t)row * D_;
  float s = 0.0f;
#pragma unroll
  for (int t = 0; t < 6; ++t) {
    const float4 q = *(const float4*)(p + lane * 4 + t * 128);
    s += q.x * q.x + q.y * q.y + q.z * q.z + q.w * q.w;
  }
#pragma unroll
  for (int m = 16; m >= 1; m >>= 1) s += __shfl_xor(s, m, 32);
  if (lane == 0) invn[row] = 1.0f / fmaxf(sqrtf(s), 1e-12f);
}

// ---------------------------------------------------------------------------
// P2 kernel 2 (fallback): convert-while-staging Gram (proven loop shape).
// ---------------------------------------------------------------------------
__global__ __launch_bounds__(256) void cpl_gram_loss_kernel(
    const float* __restrict__ feat,
    const int*   __restrict__ pos32,
    const float* __restrict__ invn,
    float* __restrict__ partial) {
  __shared__ _Float16 As[IT * SROW];
  __shared__ _Float16 Bs[JC * SROW];
  __shared__ int      iposs[IT];
  __shared__ float    rowacc[2 * IT];

  const int b    = blockIdx.x / (N_ / IT);
  const int i0   = (blockIdx.x % (N_ / IT)) * IT;
  const int tid  = threadIdx.x;
  const int wv   = tid >> 5;
  const int lane = tid & 31;
  const int isub = wv & 3;
  const int jsel = wv >> 2;
  const int l15  = lane & 15;
  const int h    = lane >> 4;

  if (tid < IT) iposs[tid] = pos32[2 * (b * N_ + i0 + tid)];

  const int   sr   = tid >> 2;
  const int   sc   = (tid & 3) * 32;
  const float invA = invn[b * N_ + i0 + sr];

  __syncthreads();

  int prow[8];
#pragma unroll
  for (int v = 0; v < 8; ++v) prow[v] = iposs[isub * 16 + 8 * h + v];
  const int igBase = i0 + isub * 16 + 8 * h;

  float racc[8];
#pragma unroll
  for (int v = 0; v < 8; ++v) racc[v] = 0.0f;

  for (int jc = 0; jc < NJC; ++jc) {
    const int   j0   = jc * JC;
    const float invB = invn[b * N_ + j0 + sr];
    const int   jg0  = j0 + jsel * 32 + l15;
    const int   jg1  = jg0 + 16;
    const int   pj0  = pos32[2 * (b * N_ + jg0)];
    const int   pj1  = pos32[2 * (b * N_ + jg1)];

    v8f acc0 = {};
    v8f acc1 = {};

    for (int kc = 0; kc < NKC; ++kc) {
      __syncthreads();
      {
        const float* src = feat + (size_t)(b * N_ + i0 + sr) * D_ + kc * KC + sc;
        _Float16* dst = &As[sr * SROW + sc];
#pragma unroll
        for (int t = 0; t < 4; ++t) {
          const float4 qa = *(const float4*)(src + t * 8);
          const float4 qb = *(const float4*)(src + t * 8 + 4);
          v8h hh;
          hh[0] = (_Float16)(qa.x * invA); hh[1] = (_Float16)(qa.y * invA);
          hh[2] = (_Float16)(qa.z * invA); hh[3] = (_Float16)(qa.w * invA);
          hh[4] = (_Float16)(qb.x * invA); hh[5] = (_Float16)(qb.y * invA);
          hh[6] = (_Float16)(qb.z * invA); hh[7] = (_Float16)(qb.w * invA);
          *(v8h*)(dst + t * 8) = hh;
        }
      }
      {
        const float* src = feat + (size_t)(b * N_ + j0 + sr) * D_ + kc * KC + sc;
        _Float16* dst = &Bs[sr * SROW + sc];
#pragma unroll
        for (int t = 0; t < 4; ++t) {
          const float4 qa = *(const float4*)(src + t * 8);
          const float4 qb = *(const float4*)(src + t * 8 + 4);
          v8h hh;
          hh[0] = (_Float16)(qa.x * invB); hh[1] = (_Float16)(qa.y * invB);
          hh[2] = (_Float16)(qa.z * invB); hh[3] = (_Float16)(qa.w * invB);
          hh[4] = (_Float16)(qb.x * invB); hh[5] = (_Float16)(qb.y * invB);
          hh[6] = (_Float16)(qb.z * invB); hh[7] = (_Float16)(qb.w * invB);
          *(v8h*)(dst + t * 8) = hh;
        }
      }
      __syncthreads();

#pragma unroll
      for (int ks = 0; ks < 4; ++ks) {
        const int kb = ks * 32;
        v16h a;
        {
          const _Float16* ar = &As[(isub * 16 + l15) * SROW + kb];
          const v8h c0 = *(const v8h*)(ar + 8 * h);
          const v8h c1 = *(const v8h*)(ar + 16 + 8 * h);
#pragma unroll
          for (int e = 0; e < 8; ++e) { a[e] = c0[e]; a[8 + e] = c1[e]; }
        }
        v16h b0f, b1f;
        {
          const _Float16* br0 = &Bs[(jsel * 32 + l15) * SROW + kb + 8 * h];
          const v8h d0 = *(const v8h*)(br0);
          const v8h d1 = *(const v8h*)(br0 + 16);
          const _Float16* br1 = &Bs[(jsel * 32 + 16 + l15) * SROW + kb + 8 * h];
          const v8h e0 = *(const v8h*)(br1);
          const v8h e1 = *(const v8h*)(br1 + 16);
#pragma unroll
          for (int e = 0; e < 8; ++e) {
            b0f[e] = d0[e]; b0f[8 + e] = d1[e];
            b1f[e] = e0[e]; b1f[8 + e] = e1[e];
          }
        }
        acc0 = __builtin_amdgcn_wmma_f32_16x16x32_f16(
            false, a, false, b0f, (short)0, acc0, false, false);
        acc1 = __builtin_amdgcn_wmma_f32_16x16x32_f16(
            false, a, false, b1f, (short)0, acc1, false, false);
      }
    }

#pragma unroll
    for (int v = 0; v < 8; ++v) {
      const int ig = igBase + v;
      float w0 = (prow[v] == pj0) ? -1.0f : 1.0f;
      w0 = (ig == jg0) ? 0.0f : w0;
      float w1 = (prow[v] == pj1) ? -1.0f : 1.0f;
      w1 = (ig == jg1) ? 0.0f : w1;
      racc[v] = fmaf(acc0[v], w0, racc[v]);
      racc[v] = fmaf(acc1[v], w1, racc[v]);
    }
  }

#pragma unroll
  for (int v = 0; v < 8; ++v) {
#pragma unroll
    for (int m = 8; m >= 1; m >>= 1) racc[v] += __shfl_xor(racc[v], m, 32);
  }
  if (l15 == 0) {
#pragma unroll
    for (int v = 0; v < 8; ++v)
      rowacc[jsel * IT + isub * 16 + 8 * h + v] = racc[v];
  }
  __syncthreads();

  if (tid < IT) {
    const float acc = rowacc[tid] + rowacc[IT + tid];
    const float x   = acc * INV_TEMP;
    const float lp  = (x > 0.0f) ? (x + log1pf(expf(-x))) : log1pf(expf(x));
    rowacc[tid] = lp;
  }
  __syncthreads();
  if (tid == 0) {
    float s = 0.0f;
    for (int r = 0; r < IT; ++r) s += rowacc[r];
    partial[blockIdx.x] = s;
  }
}

// ---------------------------------------------------------------------------
// Kernel 3: fixed-order reduction of the 576 block partials -> mean.
// ---------------------------------------------------------------------------
__global__ __launch_bounds__(256) void cpl_finalize_kernel(
    const float* __restrict__ partial, float* __restrict__ out) {
  __shared__ float sm[8];
  float s = 0.0f;
  for (int i = threadIdx.x; i < NBLK_GRAM; i += 256) s += partial[i];
#pragma unroll
  for (int m = 16; m >= 1; m >>= 1) s += __shfl_xor(s, m, 32);
  if ((threadIdx.x & 31) == 0) sm[threadIdx.x >> 5] = s;
  __syncthreads();
  if (threadIdx.x == 0) {
    float t = 0.0f;
    for (int w = 0; w < 8; ++w) t += sm[w];
    out[0] = t * (1.0f / (float)(B_ * N_));
  }
}

// ---------------------------------------------------------------------------
extern "C" void kernel_launch(void* const* d_in, const int* in_sizes, int n_in,
                              void* d_out, int out_size, void* d_ws, size_t ws_size,
                              hipStream_t stream) {
  const float* feat  = (const float*)d_in[0];    // [B, N, D] fp32
  const int*   pos32 = (const int*)d_in[1];      // [B, N] int64, LE low dwords
  float*       out   = (float*)d_out;            // scalar fp32
  (void)in_sizes; (void)n_in; (void)out_size;

  float* partial = (float*)d_ws;                 // NBLK_GRAM floats
  const size_t fhat_off = ((size_t)NBLK_GRAM * 4 + 15) & ~(size_t)15;
  const size_t need_p1  = fhat_off + (size_t)B_ * N_ * D_ * 2;

  if (ws_size >= need_p1) {
    // Preferred path: pre-normalized f16 f̂ in workspace (L2-resident),
    // async LDS staging, zero conversion VALU in the GEMM steady state.
    _Float16* fhat = (_Float16*)((char*)d_ws + fhat_off);
    cpl_normconv_kernel<<<(B_ * N_) / 8, 256, 0, stream>>>(feat, fhat);
    cpl_gram_loss_f16_kernel<<<NBLK_GRAM, 256, 0, stream>>>(fhat, pos32, partial);
  } else {
    // Fallback: ~150 KB workspace, convert while staging.
    float* invn = partial + NBLK_GRAM;
    cpl_norms_kernel<<<(B_ * N_) / 8, 256, 0, stream>>>(feat, invn);
    cpl_gram_loss_kernel<<<NBLK_GRAM, 256, 0, stream>>>(feat, pos32, invn, partial);
  }
  cpl_finalize_kernel<<<1, 256, 0, stream>>>(partial, out);
}
